// PointManifold_NNML_partseg_pointchd_32298154066460
// MI455X (gfx1250) — matrix operand order repaired and checked by reading, not compiled, not run on hardware
//
#include <hip/hip_runtime.h>

typedef __attribute__((ext_vector_type(16))) _Float16 v16h;
typedef __attribute__((ext_vector_type(8)))  float    v8f;

#define DEV static __device__ __forceinline__

#if __has_builtin(__builtin_amdgcn_global_load_async_to_lds_b128)
#define HAS_ASYNC_LDS 1
typedef __attribute__((__vector_size__(4 * sizeof(int)))) int async_v4i;
#endif

// 16-byte global -> LDS copy; async path uses the CDNA5 async-DMA instruction.
DEV void copy16_lds(void* lds, const void* g) {
#ifdef HAS_ASYNC_LDS
  __builtin_amdgcn_global_load_async_to_lds_b128(
      (async_v4i __attribute__((address_space(1)))*)(unsigned long long)g,
      (async_v4i __attribute__((address_space(3)))*)(unsigned)(unsigned long long)lds,
      0, 0);
#else
  *(int4*)lds = *(const int4*)g;
#endif
}
DEV void copy16_wait() {
#ifdef HAS_ASYNC_LDS
#if __has_builtin(__builtin_amdgcn_s_wait_asynccnt)
  __builtin_amdgcn_s_wait_asynccnt(0);
#else
  asm volatile("s_wait_asynccnt 0" ::: "memory");
#endif
#endif
}

DEV int lane_id() { return threadIdx.x & 31; }

DEV v8f wmma_f16(v16h a, v16h b, v8f c) {
  return __builtin_amdgcn_wmma_f32_16x16x32_f16(
      false, a, false, b, (short)0, c, false, false);
}

// A-tile (16x32): lane l<16 -> row l, K = {hb..hb+7, 16+hb..16+hb+7}, hb = 0/8
template <typename F>
DEV v16h make_a(F&& f) {
  const int l = lane_id();
  const int r = l & 15;
  const int hb = (l >> 4) ? 8 : 0;
  v16h a;
#pragma unroll
  for (int i = 0; i < 8; ++i) {
    a[i]     = f(r, hb + i);
    a[i + 8] = f(r, 16 + hb + i);
  }
  return a;
}
// B-tile (32x16): lane l<16 -> col l, K = {kb..kb+7, 16+kb..16+kb+7}, kb = 0/8
template <typename F>
DEV v16h make_b(F&& f) {
  const int l = lane_id();
  const int c = l & 15;
  const int kb = (l >> 4) ? 8 : 0;
  v16h b;
#pragma unroll
  for (int i = 0; i < 8; ++i) {
    b[i]     = f(kb + i, c);
    b[i + 8] = f(16 + kb + i, c);
  }
  return b;
}
// D element m of lane l: row = m + (l>=16 ? 8 : 0), col = l & 15

DEV float lrelu(float v) { return v > 0.f ? v : 0.2f * v; }
DEV float bn_scale(float g) { return g * rsqrtf(1.f + 1e-5f); }

DEV unsigned fenc(float f) {
  unsigned b = __float_as_uint(f);
  return b ^ ((unsigned)((int)b >> 31) | 0x80000000u);
}
DEV float fdec(unsigned u) {
  unsigned b = (u & 0x80000000u) ? (u ^ 0x80000000u) : ~u;
  return __uint_as_float(b);
}

DEV void wave_argmax(float& v, int& i) {
#pragma unroll
  for (int m = 16; m >= 1; m >>= 1) {
    float ov = __shfl_xor(v, m, 32);
    int   oi = __shfl_xor(i, m, 32);
    if (ov > v || (ov == v && oi < i)) { v = ov; i = oi; }
  }
}

// ---------------------------------------------------------------- stage 0
// h0 stored point-major (B, N, 32) f16, channels 9..31 zero-padded so every
// 32-wide K-chunk is fully valid (no predicated loads anywhere downstream).
__global__ void __launch_bounds__(256) build_h0(
    const float* __restrict__ x,
    const float* __restrict__ w00, const float* __restrict__ g00, const float* __restrict__ b00,
    const float* __restrict__ w01, const float* __restrict__ g01, const float* __restrict__ b01,
    _Float16* __restrict__ h0h, int B, int N) {
  int i = blockIdx.x * blockDim.x + threadIdx.x;
  if (i >= B * N) return;
  int b = i / N, n = i % N;
  const float* xb = x + (size_t)b * 3 * N;
  float X = xb[0 * N + n], Y = xb[1 * N + n], Z = xb[2 * N + n];
  auto cb = [&](float a0, float a1, const float* W, const float* G, const float* Bi,
                float& o0, float& o1) {
    float y0 = W[0] * a0 + W[1] * a1;
    float y1 = W[2] * a0 + W[3] * a1;
    o0 = lrelu(y0 * bn_scale(G[0]) + Bi[0]);
    o1 = lrelu(y1 * bn_scale(G[1]) + Bi[1]);
  };
  float z0, z1, y0, y1, x0, x1v;
  cb(X, Y, w00, g00, b00, z0, z1);   z0 *= Z;  z1 *= Z;
  cb(X, Z, w01, g01, b01, y0, y1);   y0 *= Y;  y1 *= Y;
  cb(Y, Z, w01, g01, b01, x0, x1v);  x0 *= X;  x1v *= X;
  _Float16* hb = h0h + ((size_t)b * N + n) * 32;
  hb[0] = (_Float16)X;  hb[1] = (_Float16)Y;   hb[2] = (_Float16)Z;
  hb[3] = (_Float16)x0; hb[4] = (_Float16)x1v;
  hb[5] = (_Float16)y0; hb[6] = (_Float16)y1;
  hb[7] = (_Float16)z0; hb[8] = (_Float16)z1;
#pragma unroll
  for (int c = 9; c < 32; ++c) hb[c] = (_Float16)0.f;
}

// ---------------------------------------------------------------- row norms
// Per-point squared norm over C f16 channels -> f32 (computed once, reused by
// every KNN tile instead of a 64-deep dependent FMA chain per candidate).
template <int C>
__global__ void __launch_bounds__(256) row_norm_kernel(
    const _Float16* __restrict__ xin, int cs, int off,
    float* __restrict__ nrm, int B, int N) {
  int i = blockIdx.x * blockDim.x + threadIdx.x;
  if (i >= B * N) return;
  const _Float16* r = xin + (size_t)i * cs + off;
  float s = 0.f;
#pragma unroll
  for (int c = 0; c < C; ++c) { float v = (float)r[c]; s += v * v; }
  nrm[i] = s;
}

// ---------------------------------------------------------------- KNN
// Input point-major (N, cs) f16, channels [off, off+C), C a multiple of 32 so
// every operand chunk is an unconditional contiguous load. Block = 16 queries.
template <int C>
__global__ void __launch_bounds__(512) knn_kernel(
    const _Float16* __restrict__ xin, int cs, int off,
    const float* __restrict__ nrm,
    int* __restrict__ idx_out, int N) {
  constexpr int KT = C / 32;
  const int b = blockIdx.y;
  const int q0 = blockIdx.x * 16;
  const int tid = threadIdx.x;
  const int wave = tid >> 5, lane = tid & 31;

  __shared__ float pd[16][512];
  __shared__ float pool_v[16][168];
  __shared__ int   pool_i[16][168];
  __shared__ float qn[16];

  auto row = [&](int n) { return xin + ((size_t)b * N + n) * cs + off; };

  if (tid < 16) qn[tid] = nrm[(size_t)b * N + q0 + tid];
  __syncthreads();

  v16h A[KT];
  {
    const _Float16* r = row(q0 + (lane & 15));
    for (int ks = 0; ks < KT; ++ks)
      A[ks] = make_a([&](int, int k) -> _Float16 { return r[ks * 32 + k]; });
  }

  for (int chunk = 0; chunk < 4; ++chunk) {
    const int cbase = chunk * 512;
    for (int s = wave; s < 32; s += 16) {
      const int cb = cbase + s * 16;
      const int col = lane & 15;
      const _Float16* cr = row(cb + col);
      const float cn = nrm[(size_t)b * N + cb + col];
      v8f acc = {};
      for (int ks = 0; ks < KT; ++ks) {
        v16h Bt = make_b([&](int k, int) -> _Float16 { return cr[ks * 32 + k]; });
        acc = wmma_f16(A[ks], Bt, acc);
      }
      const int rb = (lane >= 16) ? 8 : 0;
#pragma unroll
      for (int m = 0; m < 8; ++m)
        pd[rb + m][s * 16 + col] = 2.f * acc[m] - qn[rb + m] - cn;
    }
    __syncthreads();
    {
      float* prow = pd[wave];
      for (int r = 0; r < 40; ++r) {
        float bv = -3.4e38f; int bi = 0x7FFFFFFF;
        for (int e = lane; e < 512; e += 32) {
          float v = prow[e];
          if (v > bv || (v == bv && e < bi)) { bv = v; bi = e; }
        }
        wave_argmax(bv, bi);
        if (lane == 0) {
          pool_v[wave][chunk * 40 + r] = bv;
          pool_i[wave][chunk * 40 + r] = cbase + bi;
          prow[bi] = -3.4e38f;
        }
      }
    }
    __syncthreads();
  }
  for (int r = 0; r < 40; ++r) {
    float bv = -3.4e38f; int bs = 0x7FFFFFFF;
    for (int e = lane; e < 160; e += 32) {
      float v = pool_v[wave][e];
      if (v > bv || (v == bv && e < bs)) { bv = v; bs = e; }
    }
    wave_argmax(bv, bs);
    if (lane == 0) {
      idx_out[((size_t)b * N + q0 + wave) * 40 + r] = pool_i[wave][bs];
      pool_v[wave][bs] = -3.4e38f;
    }
  }
}

// ---------------------------------------------------------------- edge conv
// C = real channels, CP = padded channels. Edge feature layout: [0,CP) =
// feat-center, [CP,2CP) = center; zero-padding for C<CP is folded into the W1
// staging remap so operand assembly has no per-element predicates. K padded to
// 48 neighbours by repeating neighbour 0 (max-invariant).
template <int C, int CP, int CO1, int CO2>
__global__ void __launch_bounds__(256) edgeconv_kernel(
    const _Float16* __restrict__ xin, int cs_in, int off_in,
    const int* __restrict__ knn_idx,
    const float* __restrict__ w1, const float* __restrict__ g1, const float* __restrict__ b1,
    const float* __restrict__ w2, const float* __restrict__ g2, const float* __restrict__ b2,
    _Float16* __restrict__ xout, int cs_out, int off_out, int N) {
  constexpr int E = 2 * CP;
  constexpr int KT1 = E / 32;
  constexpr int CT1 = CO1 / 16;
  constexpr int CO = (CO2 > 0) ? CO2 : CO1;
  constexpr int CO2S = (CO2 > 0) ? CO2 : 1;
  constexpr int MIDW = (CO2 > 0) ? CO1 : 1;

  const int b = blockIdx.y, p0 = blockIdx.x * 16;
  const int tid = threadIdx.x, wave = tid >> 5, lane = tid & 31;
  const int col = lane & 15, rb = (lane >= 16) ? 8 : 0;

  __shared__ _Float16 centerS[16][CP];
  __shared__ int idxS[16][48];
  __shared__ _Float16 w1S[CO1][E];
  __shared__ _Float16 w2S[CO2S][MIDW];
  __shared__ _Float16 midS[8][16][MIDW];
  __shared__ float sc1S[CO1], bi1S[CO1];
  __shared__ float sc2S[CO2S], bi2S[CO2S];
  __shared__ unsigned pmaxS[16][CO];

  // async-stage neighbour lists: 40 ints/point = 10 x b128
  for (int i = tid; i < 16 * 10; i += 256)
    copy16_lds(&idxS[i / 10][(i % 10) * 4],
               &knn_idx[((size_t)b * N + p0 + i / 10) * 40 + (i % 10) * 4]);
  for (int i = tid; i < 16 * CP; i += 256)
    centerS[i / CP][i % CP] =
        xin[((size_t)b * N + p0 + i / CP) * cs_in + off_in + (i % CP)];
  // W1 remap into padded edge-channel space: [0,CP) diff part, [CP,2CP) center
  for (int i = tid; i < CO1 * E; i += 256) {
    int o = i / E, cp = i % E;
    int half = (cp >= CP) ? 1 : 0;
    int cc = cp - half * CP;
    w1S[o][cp] = (cc < C) ? (_Float16)w1[o * (2 * C) + half * C + cc] : (_Float16)0.f;
  }
  for (int i = tid; i < CO1; i += 256) { sc1S[i] = bn_scale(g1[i]); bi1S[i] = b1[i]; }
  if constexpr (CO2 > 0) {
    for (int i = tid; i < CO2 * CO1; i += 256) w2S[i / CO1][i % CO1] = (_Float16)w2[i];
    for (int i = tid; i < CO2; i += 256) { sc2S[i] = bn_scale(g2[i]); bi2S[i] = b2[i]; }
  }
  for (int i = tid; i < 16 * CO; i += 256) pmaxS[i / CO][i % CO] = 0u;
  copy16_wait();
  __syncthreads();
  for (int i = tid; i < 16 * 8; i += 256) idxS[i / 8][40 + (i % 8)] = idxS[i / 8][0];
  __syncthreads();

  for (int pt = wave; pt < 48; pt += 8) {
    const int p = pt / 3, jb = (pt % 3) * 16;
    const int q = idxS[p][jb + col];                 // this lane's A-row neighbour
    const _Float16* qrow = xin + ((size_t)b * N + q) * cs_in + off_in;

    v16h A1[KT1];
    for (int ks = 0; ks < KT1; ++ks)
      A1[ks] = make_a([&](int, int k) -> _Float16 {
        int c = ks * 32 + k;
        if (c < CP) return (_Float16)(qrow[c] - centerS[p][c]);
        return centerS[p][c - CP];
      });

    for (int ct = 0; ct < CT1; ++ct) {
      v8f acc = {};
      for (int ks = 0; ks < KT1; ++ks) {
        v16h Bt = make_b([&](int k, int c) -> _Float16 { return w1S[ct * 16 + c][ks * 32 + k]; });
        acc = wmma_f16(A1[ks], Bt, acc);
      }
      const int ch = ct * 16 + col;
      const float sc = sc1S[ch], bi = bi1S[ch];
      if constexpr (CO2 > 0) {
#pragma unroll
        for (int m = 0; m < 8; ++m)
          midS[wave][rb + m][ch] = (_Float16)lrelu(acc[m] * sc + bi);
      } else {
        float mx = -3.4e38f;
#pragma unroll
        for (int m = 0; m < 8; ++m) mx = fmaxf(mx, lrelu(acc[m] * sc + bi));
        mx = fmaxf(mx, __shfl_xor(mx, 16, 32));
        if (lane < 16) atomicMax(&pmaxS[p][ch], fenc(mx));
      }
    }
    if constexpr (CO2 > 0) {
      constexpr int KT2 = CO1 / 32, CT2 = CO2 / 16;
      v16h A2[KT2];
      for (int ks = 0; ks < KT2; ++ks)
        A2[ks] = make_a([&](int r, int k) -> _Float16 { return midS[wave][r][ks * 32 + k]; });
      for (int ct = 0; ct < CT2; ++ct) {
        v8f acc = {};
        for (int ks = 0; ks < KT2; ++ks) {
          v16h Bt = make_b([&](int k, int c) -> _Float16 { return w2S[ct * 16 + c][ks * 32 + k]; });
          acc = wmma_f16(A2[ks], Bt, acc);
        }
        const int ch = ct * 16 + col;
        const float sc = sc2S[ch], bi = bi2S[ch];
        float mx = -3.4e38f;
#pragma unroll
        for (int m = 0; m < 8; ++m) mx = fmaxf(mx, lrelu(acc[m] * sc + bi));
        mx = fmaxf(mx, __shfl_xor(mx, 16, 32));
        if (lane < 16) atomicMax(&pmaxS[p][ch], fenc(mx));
      }
    }
  }
  __syncthreads();
  for (int i = tid; i < 16 * CO; i += 256)
    xout[((size_t)b * N + p0 + i / CO) * cs_out + off_out + (i % CO)] =
        (_Float16)fdec(pmaxS[i / CO][i % CO]);
}

// ---------------------------------------------------------------- pointwise conv
// A = activations (16 points x CIN, point-major f16), B = W^T staged in LDS.
// SRC: 0 = in0 rows of CIN; 2 = concat[g(1024) broadcast, xcat(192)]
// MODE: 0 = BN+LReLU -> f16 (N,COUT); 1 = BN+LReLU -> global max into gkeys;
//       2 = plain f32 store channel-major, rows guarded by cout_real
template <int CIN, int COUT, int SRC, int MODE>
__global__ void __launch_bounds__(256) conv_kernel(
    const _Float16* __restrict__ in0, const _Float16* __restrict__ in1,
    const float* __restrict__ w, const float* __restrict__ gp, const float* __restrict__ bp,
    _Float16* __restrict__ out16, float* __restrict__ outf,
    unsigned* __restrict__ gkeys, int N, int cout_real) {
  constexpr int KT = CIN / 32, OT = COUT / 16;
  const int tid = threadIdx.x, wave = tid >> 5, lane = tid & 31;
  const int col = lane & 15, rb = (lane >= 16) ? 8 : 0;
  const int ntb = N / (16 * 8);                    // blocks per (b, ot)
  const int bo = blockIdx.x / ntb;
  const int b = bo / OT, ot = bo % OT;
  const int n0 = ((blockIdx.x % ntb) * 8 + wave) * 16;

  __shared__ _Float16 wS[16][CIN];
  __shared__ float scS[16], biS[16];

  for (int i = tid; i < 16 * CIN; i += 256) {
    int o = i / CIN, c = i % CIN;
    int og = ot * 16 + o;
    if ((i & 63) == 0) __builtin_prefetch(&w[(size_t)og * CIN + c + 256], 0, 0);
    wS[o][c] = (og < cout_real) ? (_Float16)w[(size_t)og * CIN + c] : (_Float16)0.f;
  }
  if constexpr (MODE != 2) {
    if (tid < 16) { scS[tid] = bn_scale(gp[ot * 16 + tid]); biS[tid] = bp[ot * 16 + tid]; }
  }
  __syncthreads();

  auto src = [&](int n, int c) -> _Float16 {
    if constexpr (SRC == 0) {
      return in0[((size_t)b * N + n) * CIN + c];
    } else {
      if (c < 1024) return in0[(size_t)b * 1024 + c];
      return in1[((size_t)b * N + n) * 192 + (c - 1024)];
    }
  };

  v8f acc = {};
  for (int ks = 0; ks < KT; ++ks) {
    v16h At = make_a([&](int r, int k) -> _Float16 { return src(n0 + r, ks * 32 + k); });
    v16h Bt = make_b([&](int k, int c) -> _Float16 { return wS[c][ks * 32 + k]; });
    acc = wmma_f16(At, Bt, acc);
  }
  const int ch = ot * 16 + col;
  if constexpr (MODE == 1) {
    float mx = -3.4e38f;
#pragma unroll
    for (int m = 0; m < 8; ++m) mx = fmaxf(mx, lrelu(acc[m] * scS[col] + biS[col]));
    atomicMax(&gkeys[(size_t)b * COUT + ch], fenc(mx));
  } else if constexpr (MODE == 0) {
#pragma unroll
    for (int m = 0; m < 8; ++m)
      out16[((size_t)b * N + n0 + rb + m) * COUT + ch] =
          (_Float16)lrelu(acc[m] * scS[col] + biS[col]);
  } else {
    if (ch < cout_real) {
#pragma unroll
      for (int m = 0; m < 8; ++m)
        outf[((size_t)b * cout_real + ch) * N + n0 + rb + m] = acc[m];
    }
  }
}

__global__ void init_gkeys(unsigned* g, int n) {
  int i = blockIdx.x * blockDim.x + threadIdx.x;
  if (i < n) g[i] = 0u;
}
__global__ void decode_g(const unsigned* k, _Float16* g, int n) {
  int i = blockIdx.x * blockDim.x + threadIdx.x;
  if (i < n) g[i] = (_Float16)fdec(k[i]);
}

// ---------------------------------------------------------------- launch
extern "C" void kernel_launch(void* const* d_in, const int* in_sizes, int n_in,
                              void* d_out, int out_size, void* d_ws, size_t ws_size,
                              hipStream_t stream) {
  (void)in_sizes; (void)n_in; (void)out_size; (void)ws_size;
  const int B = 8, N = 2048;
  const float* x   = (const float*)d_in[0];
  const float* w00 = (const float*)d_in[1];
  const float* g00 = (const float*)d_in[2];
  const float* b00 = (const float*)d_in[3];
  const float* w01 = (const float*)d_in[4];
  const float* g01 = (const float*)d_in[5];
  const float* b01 = (const float*)d_in[6];
  const float *W1 = (const float*)d_in[7],  *G1 = (const float*)d_in[8],  *B1 = (const float*)d_in[9];
  const float *W2 = (const float*)d_in[10], *G2 = (const float*)d_in[11], *B2 = (const float*)d_in[12];
  const float *W3 = (const float*)d_in[13], *G3 = (const float*)d_in[14], *B3 = (const float*)d_in[15];
  const float *W4 = (const float*)d_in[16], *G4 = (const float*)d_in[17], *B4 = (const float*)d_in[18];
  const float *W5 = (const float*)d_in[19], *G5 = (const float*)d_in[20], *B5 = (const float*)d_in[21];
  const float *W6 = (const float*)d_in[22], *G6 = (const float*)d_in[23], *B6 = (const float*)d_in[24];
  const float *W7 = (const float*)d_in[25], *G7 = (const float*)d_in[26], *B7 = (const float*)d_in[27];
  const float *W8 = (const float*)d_in[28], *G8 = (const float*)d_in[29], *B8 = (const float*)d_in[30];
  const float *W9 = (const float*)d_in[31], *G9 = (const float*)d_in[32], *B9 = (const float*)d_in[33];
  const float* W10 = (const float*)d_in[34];
  float* out = (float*)d_out;

  char* ws = (char*)d_ws;
  size_t off = 0;
  auto alloc = [&](size_t bytes) -> void* {
    void* p = ws + off;
    off += (bytes + 255) & ~(size_t)255;
    return p;
  };
  _Float16* h0h  = (_Float16*)alloc((size_t)B * N * 32 * 2);
  int* idx1 = (int*)alloc((size_t)B * N * 40 * 4);
  int* idx2 = (int*)alloc((size_t)B * N * 40 * 4);
  int* idx3 = (int*)alloc((size_t)B * N * 40 * 4);
  _Float16* xcat = (_Float16*)alloc((size_t)B * N * 192 * 2);  // [x1|x2|x3]
  float* nrmbuf  = (float*)alloc((size_t)B * N * 4);
  unsigned* gkeys = (unsigned*)alloc((size_t)B * 1024 * 4);
  _Float16* g16  = (_Float16*)alloc((size_t)B * 1024 * 2);
  _Float16* h7   = (_Float16*)alloc((size_t)B * N * 256 * 2);
  _Float16* h8   = (_Float16*)alloc((size_t)B * N * 256 * 2);
  _Float16* h9   = (_Float16*)alloc((size_t)B * N * 128 * 2);

  const dim3 tiles(N / 16, B);
  const int npts = (B * N + 255) / 256;

  build_h0<<<npts, 256, 0, stream>>>(
      x, w00, g00, b00, w01, g01, b01, h0h, B, N);

  row_norm_kernel<32><<<npts, 256, 0, stream>>>(h0h, 32, 0, nrmbuf, B, N);
  knn_kernel<32><<<tiles, 512, 0, stream>>>(h0h, 32, 0, nrmbuf, idx1, N);
  edgeconv_kernel<9, 16, 64, 64><<<tiles, 256, 0, stream>>>(
      h0h, 32, 0, idx1, W1, G1, B1, W2, G2, B2, xcat, 192, 0, N);

  row_norm_kernel<64><<<npts, 256, 0, stream>>>(xcat, 192, 0, nrmbuf, B, N);
  knn_kernel<64><<<tiles, 512, 0, stream>>>(xcat, 192, 0, nrmbuf, idx2, N);
  edgeconv_kernel<64, 64, 64, 64><<<tiles, 256, 0, stream>>>(
      xcat, 192, 0, idx2, W3, G3, B3, W4, G4, B4, xcat, 192, 64, N);

  row_norm_kernel<64><<<npts, 256, 0, stream>>>(xcat, 192, 64, nrmbuf, B, N);
  knn_kernel<64><<<tiles, 512, 0, stream>>>(xcat, 192, 64, nrmbuf, idx3, N);
  edgeconv_kernel<64, 64, 64, 0><<<tiles, 256, 0, stream>>>(
      xcat, 192, 64, idx3, W5, G5, B5, nullptr, nullptr, nullptr, xcat, 192, 128, N);

  init_gkeys<<<(B * 1024 + 255) / 256, 256, 0, stream>>>(gkeys, B * 1024);
  conv_kernel<192, 1024, 0, 1><<<B * 64 * (N / 128), 256, 0, stream>>>(
      xcat, nullptr, W6, G6, B6, nullptr, nullptr, gkeys, N, 1024);
  decode_g<<<(B * 1024 + 255) / 256, 256, 0, stream>>>(gkeys, g16, B * 1024);

  conv_kernel<1216, 256, 2, 0><<<B * 16 * (N / 128), 256, 0, stream>>>(
      g16, xcat, W7, G7, B7, h7, nullptr, nullptr, N, 256);
  conv_kernel<256, 256, 0, 0><<<B * 16 * (N / 128), 256, 0, stream>>>(
      h7, nullptr, W8, G8, B8, h8, nullptr, nullptr, N, 256);
  conv_kernel<256, 128, 0, 0><<<B * 8 * (N / 128), 256, 0, stream>>>(
      h8, nullptr, W9, G9, B9, h9, nullptr, nullptr, N, 128);
  conv_kernel<128, 64, 0, 2><<<B * 4 * (N / 128), 256, 0, stream>>>(
      h9, nullptr, W10, nullptr, nullptr, nullptr, out, nullptr, N, 50);
}
